// TransformerBlock_24335284699524
// MI455X (gfx1250) — matrix operand order repaired and checked by reading, not compiled, not run on hardware
//
#include <hip/hip_runtime.h>
#include <hip/hip_bf16.h>

typedef __attribute__((ext_vector_type(16))) _Float16 v16h;
typedef __attribute__((ext_vector_type(8)))  _Float16 v8h;
typedef __attribute__((ext_vector_type(8)))  float    v8f;

#define NPIX (1024*1024)
#define EPS 1e-5f
#define SCALE 0.17677669529663687f

#define WMMA_F16(a,b,c) __builtin_amdgcn_wmma_f32_16x16x32_f16(false,(a),false,(b),(short)0,(c),false,false)

__device__ __forceinline__ float gelu_erf(float x) {
  return 0.5f * x * (1.0f + erff(x * 0.70710678118654752f));
}

// ---------------------------------------------------------------------------
// Depthwise 3x3 conv on [8,1024,1024] + per-channel stats (GN groups==C).
// One thread per pixel, all 8 channels -> y stored pixel-major [pix][8] f16
// so the branch convs can fetch all 8 input channels with one b128 load.
// ---------------------------------------------------------------------------
__global__ void k_dwconv(const float* __restrict__ xp, const float* __restrict__ dww,
                         _Float16* __restrict__ y, float* __restrict__ stats) {
  __shared__ float sAcc[16];
  int tid = threadIdx.x;
  int lane = tid & 31;
  if (tid < 16) sAcc[tid] = 0.f;
  __syncthreads();
  int p = blockIdx.x * 256 + tid;
  int py = p >> 10, px = p & 1023;
  v8h out;
  #pragma unroll
  for (int c = 0; c < 8; ++c) {
    float acc = 0.f;
    #pragma unroll
    for (int kh = 0; kh < 3; ++kh)
      #pragma unroll
      for (int kw = 0; kw < 3; ++kw) {
        int yy = py + kh - 1, xx = px + kw - 1;
        if (yy >= 0 && yy < 1024 && xx >= 0 && xx < 1024)
          acc += dww[c * 9 + kh * 3 + kw] * xp[(size_t)c * NPIX + yy * 1024 + xx];
      }
    out[c] = (_Float16)acc;
    float s1 = acc, s2 = acc * acc;
    #pragma unroll
    for (int d = 1; d < 32; d <<= 1) {
      s1 += __shfl_xor(s1, d, 32);
      s2 += __shfl_xor(s2, d, 32);
    }
    if (lane == 0) {
      atomicAdd(&sAcc[c * 2], s1);
      atomicAdd(&sAcc[c * 2 + 1], s2);
    }
  }
  *(v8h*)(y + (size_t)p * 8) = out;
  __syncthreads();
  if (tid < 16) atomicAdd(&stats[tid], sAcc[tid]);
}

// ---------------------------------------------------------------------------
// GN finalize: per-channel scale/shift from per-group sum/sumsq
// ---------------------------------------------------------------------------
__global__ void k_finalize(const float* __restrict__ stats, const float* __restrict__ gamma,
                           const float* __restrict__ beta, float* __restrict__ scale,
                           float* __restrict__ shift, int nCh, int chPerGroup, float invCount) {
  int c = threadIdx.x;
  if (c < nCh) {
    int g = c / chPerGroup;
    float mean = stats[2 * g] * invCount;
    float var  = stats[2 * g + 1] * invCount - mean * mean;
    float rs = rsqrtf(var + EPS);
    float sc = gamma[c] * rs;
    scale[c] = sc;
    shift[c] = beta[c] - mean * sc;
  }
}

// ---------------------------------------------------------------------------
// Fused GN-affine + GELU elementwise (in place), optional next-GN stats.
// Channel id = (gid >> cshift) & cmask  (pixel-major y: 0,7; ch-major z: 20,127)
// Stats path requires channel uniform per block (true for the z layout).
// ---------------------------------------------------------------------------
__global__ void k_gn_gelu(_Float16* __restrict__ buf, const float* __restrict__ scale,
                          const float* __restrict__ shift, float* __restrict__ statsOut,
                          int chPerGroup, int cshift, int cmask) {
  __shared__ float sh[256];
  int tid = threadIdx.x;
  int gid = blockIdx.x * 256 + tid;
  int cch = (gid >> cshift) & cmask;
  float v = (float)buf[gid] * scale[cch] + shift[cch];
  v = gelu_erf(v);
  buf[gid] = (_Float16)v;
  if (statsOut) {
    int g = cch / chPerGroup;
    sh[tid] = v; __syncthreads();
    for (int s = 128; s > 0; s >>= 1) { if (tid < s) sh[tid] += sh[tid + s]; __syncthreads(); }
    if (!tid) atomicAdd(&statsOut[g * 2], sh[0]);
    __syncthreads();
    sh[tid] = v * v; __syncthreads();
    for (int s = 128; s > 0; s >>= 1) { if (tid < s) sh[tid] += sh[tid + s]; __syncthreads(); }
    if (!tid) atomicAdd(&statsOut[g * 2 + 1], sh[0]);
  }
}

// ---------------------------------------------------------------------------
// Branch conv 8->32 ch, 3x3, dilation d, implicit GEMM with WMMA.
// Logical K ordering t = tap*8 + ci (tap-major): every 8-element fragment
// half shares ONE tap.
//   A: weights staged once per block into LDS as [tap][oc][ci] f16 with taps
//      9..11 zero-padded -> one unpredicated ds_load_b128 per group.
//   B: y is pixel-major -> one predicated global b128 load per group.
// K = 9*8 = 72, padded to 96 (3 wmma per tile).
// ---------------------------------------------------------------------------
__global__ void k_brconv(const _Float16* __restrict__ y, const float* __restrict__ W,
                         _Float16* __restrict__ z, float* __restrict__ stats,
                         int dil, int branch) {
  __shared__ _Float16 wLds[12 * 32 * 8];   // [tap][oc][ci], 6KB
  __shared__ float sAcc[64];
  int tid = threadIdx.x;
  if (tid < 64) sAcc[tid] = 0.f;
  // Stage + transpose + f16-convert weights: thread = (oc, ci), 9 contiguous f32
  {
    int oc = tid >> 3, ci = tid & 7;
    const float* wp = W + oc * 72 + ci * 9;
    #pragma unroll
    for (int tap = 0; tap < 9; ++tap)
      wLds[(tap * 32 + oc) * 8 + ci] = (_Float16)wp[tap];
    #pragma unroll
    for (int tap = 9; tap < 12; ++tap)
      wLds[(tap * 32 + oc) * 8 + ci] = (_Float16)0.f;
  }
  __syncthreads();
  int wave = blockIdx.x * 8 + (tid >> 5);
  int lane = tid & 31, ml = lane & 15, hi = lane >> 4;
  int chHalf = wave >> 16;              // 131072 waves: 2 ch-halves x 65536 tiles
  int t = wave & 65535;
  int py = t >> 6;
  int px0 = (t & 63) << 4;
  int xxb = px0 + ml;
  v8f c = {};
  #pragma unroll
  for (int kc = 0; kc < 3; ++kc) {
    // A: group g: tap = kc*4 + hi*2 + g (zero-padded, no predicate)
    v16h a;
    #pragma unroll
    for (int g = 0; g < 2; ++g) {
      int tap = kc * 4 + hi * 2 + g;
      v8h av = *(const v8h*)(&wLds[(tap * 32 + chHalf * 16 + ml) * 8]);
      #pragma unroll
      for (int ci = 0; ci < 8; ++ci)
        a[g * 8 + ci] = av[ci];
    }
    // B: im2col.  group g: tap = kc*4 + hi*2 + g; both hi-cases are constants.
    v16h b;
    #pragma unroll
    for (int g = 0; g < 2; ++g) {
      const int t0 = kc * 4 + g;            // hi == 0
      const int t1 = t0 + 2;                // hi == 1
      const int s0 = (t0 < 9) ? t0 : 0;
      const int s1 = (t1 < 9) ? t1 : 0;
      const int kh0 = s0 / 3, kw0 = s0 - kh0 * 3;   // compile-time constants
      const int kh1 = s1 / 3, kw1 = s1 - kh1 * 3;   // compile-time constants
      bool tok = hi ? (t1 < 9) : (t0 < 9);
      int dy = (hi ? kh1 : kh0) - 1;
      int dx = (hi ? kw1 : kw0) - 1;
      int yy = py + dy * dil;
      int xx = xxb + dx * dil;
      bool ok = tok && yy >= 0 && yy < 1024 && xx >= 0 && xx < 1024;
      v8h rv = {};
      if (ok) rv = *(const v8h*)(y + (size_t)(yy * 1024 + xx) * 8);
      #pragma unroll
      for (int ci = 0; ci < 8; ++ci)
        b[g * 8 + ci] = rv[ci];
    }
    c = WMMA_F16(a, b, c);
  }
  int clb = chHalf * 16 + 8 * hi;
  #pragma unroll
  for (int r = 0; r < 8; ++r) {
    float vv = c[r];
    z[(size_t)(branch * 32 + clb + r) * NPIX + py * 1024 + px0 + ml] = (_Float16)vv;
    atomicAdd(&sAcc[(clb + r) * 2], vv);
    atomicAdd(&sAcc[(clb + r) * 2 + 1], vv * vv);
  }
  __syncthreads();
  if (tid < 64) {
    int clx = tid >> 1;
    atomicAdd(&stats[16 + (branch * 32 + clx) * 2 + (tid & 1)], sAcc[tid]);
  }
}

// ---------------------------------------------------------------------------
// Final GN(32 groups of 3)+GELU folded into head 1x1 conv 96->8 per pixel
// ---------------------------------------------------------------------------
__global__ void k_head(const _Float16* __restrict__ z, const float* __restrict__ s2,
                       const float* __restrict__ sh2, const float* __restrict__ hw,
                       const float* __restrict__ hb, float* __restrict__ biasOut) {
  __shared__ float hws[768];
  int tid = threadIdx.x;
  for (int i = tid; i < 768; i += 256) hws[i] = hw[i];
  __syncthreads();
  int p = blockIdx.x * 256 + tid;
  float acc[8];
  #pragma unroll
  for (int h = 0; h < 8; ++h) acc[h] = 0.f;
  for (int c = 0; c < 96; ++c) {
    float va = gelu_erf((float)z[(size_t)c * NPIX + p] * s2[c] + sh2[c]);
    #pragma unroll
    for (int h = 0; h < 8; ++h) acc[h] += hws[h * 96 + c] * va;
  }
  #pragma unroll
  for (int h = 0; h < 8; ++h)
    biasOut[(size_t)h * NPIX + p] = acc[h] + hb[h];
}

// ---------------------------------------------------------------------------
// LayerNorm over rows of [1024,256] f32 -> f16
// ---------------------------------------------------------------------------
__global__ void k_layernorm(const float* __restrict__ x, const float* __restrict__ g,
                            const float* __restrict__ b, _Float16* __restrict__ out) {
  __shared__ float sh[256];
  __shared__ float sm, srs;
  int tid = threadIdx.x;
  int row = blockIdx.x;
  float v = x[row * 256 + tid];
  sh[tid] = v; __syncthreads();
  for (int s = 128; s > 0; s >>= 1) { if (tid < s) sh[tid] += sh[tid + s]; __syncthreads(); }
  if (!tid) sm = sh[0] * (1.f / 256.f);
  __syncthreads();
  float d = v - sm;
  sh[tid] = d * d; __syncthreads();
  for (int s = 128; s > 0; s >>= 1) { if (tid < s) sh[tid] += sh[tid + s]; __syncthreads(); }
  if (!tid) srs = rsqrtf(sh[0] * (1.f / 256.f) + EPS);
  __syncthreads();
  out[row * 256 + tid] = (_Float16)((v - sm) * srs * g[tid] + b[tid]);
}

// ---------------------------------------------------------------------------
// Generic WMMA GEMM: out = [act]( A_f16[M,K] @ W_f32[K,N] + bias ) [+ res]
// One wave per 16x16 tile.
// ---------------------------------------------------------------------------
__global__ void k_gemm16(const _Float16* __restrict__ A, const float* __restrict__ W,
                         const float* __restrict__ bias, const float* __restrict__ res,
                         float* __restrict__ oF32, _Float16* __restrict__ oF16,
                         int M, int K, int N, int act) {
  int wave = blockIdx.x * (blockDim.x >> 5) + (threadIdx.x >> 5);
  int nTn = N >> 4;
  if (wave >= (M >> 4) * nTn) return;
  int tm = wave / nTn, tn = wave - tm * nTn;
  int lane = threadIdx.x & 31, ml = lane & 15, hi = lane >> 4;
  int m0 = tm << 4, n0 = tn << 4;
  v8f c = {};
  for (int k0 = 0; k0 < K; k0 += 32) {
    v16h a, b;
    #pragma unroll
    for (int i = 0; i < 16; ++i) {
      int ka = k0 + hi * 8 + (i < 8 ? i : i + 8);
      a[i] = A[(m0 + ml) * K + ka];
      int kb = k0 + hi * 8 + (i < 8 ? i : i + 8);
      b[i] = (_Float16)W[kb * N + n0 + ml];
    }
    c = WMMA_F16(a, b, c);
  }
  int col = n0 + ml;
  #pragma unroll
  for (int r = 0; r < 8; ++r) {
    int mm = m0 + r + 8 * hi;
    float vv = c[r] + bias[col];
    if (act) vv = gelu_erf(vv);
    if (res) vv += res[mm * N + col];
    if (oF32) oF32[mm * N + col] = vv;
    if (oF16) oF16[mm * N + col] = (_Float16)vv;
  }
}

// ---------------------------------------------------------------------------
// Flash attention with additive bias + mask. One wave per (head, 16-row tile).
// D=32 == WMMA K, so q is a single A-frag; j processed 32 columns per step.
// ---------------------------------------------------------------------------
__global__ void k_attn(const _Float16* __restrict__ q, const _Float16* __restrict__ kk,
                       const _Float16* __restrict__ vv, const float* __restrict__ bias,
                       const float* __restrict__ gate, const int* __restrict__ mask,
                       _Float16* __restrict__ outp) {
  __shared__ float pT[16][32];
  int lane = threadIdx.x & 31, ml = lane & 15, hi = lane >> 4;
  int h = blockIdx.x >> 6;
  int i0 = (blockIdx.x & 63) << 4;
  v16h aq;
  #pragma unroll
  for (int i = 0; i < 16; ++i) {
    int d = hi * 8 + (i < 8 ? i : i + 8);
    aq[i] = q[(i0 + ml) * 256 + h * 32 + d];
  }
  float gh = gate[h];
  v8f o0 = {}, o1 = {};
  float Mx[8], Sm[8];
  #pragma unroll
  for (int r = 0; r < 8; ++r) { Mx[r] = -1e30f; Sm[r] = 0.f; }
  const float* bRow = bias + (size_t)h * NPIX + (size_t)i0 * 1024;
  for (int js = 0; js < 32; ++js) {
    int j0 = js * 32;
    v16h b0, b1;
    #pragma unroll
    for (int i = 0; i < 16; ++i) {
      int d = hi * 8 + (i < 8 ? i : i + 8);
      b0[i] = kk[(j0 + ml) * 256 + h * 32 + d];
      b1[i] = kk[(j0 + 16 + ml) * 256 + h * 32 + d];
    }
    v8f zc = {};
    v8f s0 = WMMA_F16(aq, b0, zc);
    v8f s1 = WMMA_F16(aq, b1, zc);
    float pad0 = mask[j0 + ml] ? 0.f : -1e9f;
    float pad1 = mask[j0 + 16 + ml] ? 0.f : -1e9f;
    #pragma unroll
    for (int r = 0; r < 8; ++r) {
      int mrow = r + 8 * hi;
      float l0 = s0[r] * SCALE + gh * bRow[mrow * 1024 + j0 + ml] + pad0;
      float l1 = s1[r] * SCALE + gh * bRow[mrow * 1024 + j0 + 16 + ml] + pad1;
      float mx = fmaxf(l0, l1);
      #pragma unroll
      for (int d = 1; d < 16; d <<= 1) mx = fmaxf(mx, __shfl_xor(mx, d, 32));
      float nM = fmaxf(Mx[r], mx);
      float alpha = __expf(Mx[r] - nM);
      float p0 = __expf(l0 - nM), p1 = __expf(l1 - nM);
      float su = p0 + p1;
      #pragma unroll
      for (int d = 1; d < 16; d <<= 1) su += __shfl_xor(su, d, 32);
      Sm[r] = Sm[r] * alpha + su;
      Mx[r] = nM;
      o0[r] *= alpha; o1[r] *= alpha;
      pT[mrow][ml] = p0;
      pT[mrow][16 + ml] = p1;
    }
    __syncthreads();
    v16h ap, bv0, bv1;
    #pragma unroll
    for (int i = 0; i < 16; ++i) {
      int kA = hi * 8 + (i < 8 ? i : i + 8);
      ap[i] = (_Float16)pT[ml][kA];
      int jl = hi * 8 + (i < 8 ? i : i + 8);
      bv0[i] = vv[(j0 + jl) * 256 + h * 32 + ml];
      bv1[i] = vv[(j0 + jl) * 256 + h * 32 + 16 + ml];
    }
    o0 = WMMA_F16(ap, bv0, o0);
    o1 = WMMA_F16(ap, bv1, o1);
    __syncthreads();
  }
  #pragma unroll
  for (int r = 0; r < 8; ++r) {
    int mrow = r + 8 * hi;
    float inv = 1.f / Sm[r];
    outp[(i0 + mrow) * 256 + h * 32 + ml]      = (_Float16)(o0[r] * inv);
    outp[(i0 + mrow) * 256 + h * 32 + 16 + ml] = (_Float16)(o1[r] * inv);
  }
}

// ---------------------------------------------------------------------------
extern "C" void kernel_launch(void* const* d_in, const int* in_sizes, int n_in,
                              void* d_out, int out_size, void* d_ws, size_t ws_size,
                              hipStream_t stream) {
  (void)in_sizes; (void)n_in; (void)out_size; (void)ws_size;
  const float* x      = (const float*)d_in[0];
  const float* xpw    = (const float*)d_in[1];
  const int*   mask   = (const int*)d_in[2];
  const float* dw_w   = (const float*)d_in[3];
  const float* dw_g   = (const float*)d_in[4];
  const float* dw_b   = (const float*)d_in[5];
  const float* post_g = (const float*)d_in[6];
  const float* post_b = (const float*)d_in[7];
  const float* head_w = (const float*)d_in[8];
  const float* head_b = (const float*)d_in[9];
  const float* ln1_g  = (const float*)d_in[10];
  const float* ln1_b  = (const float*)d_in[11];
  const float* wq = (const float*)d_in[12]; const float* bq = (const float*)d_in[13];
  const float* wk = (const float*)d_in[14]; const float* bk = (const float*)d_in[15];
  const float* wv = (const float*)d_in[16]; const float* bv = (const float*)d_in[17];
  const float* wo = (const float*)d_in[18]; const float* bo = (const float*)d_in[19];
  const float* gate  = (const float*)d_in[20];
  const float* ln2_g = (const float*)d_in[21]; const float* ln2_b = (const float*)d_in[22];
  const float* w1 = (const float*)d_in[23]; const float* b1 = (const float*)d_in[24];
  const float* w2 = (const float*)d_in[25]; const float* b2 = (const float*)d_in[26];
  const float* br_w[3] = {(const float*)d_in[27], (const float*)d_in[30], (const float*)d_in[33]};
  const float* br_g[3] = {(const float*)d_in[28], (const float*)d_in[31], (const float*)d_in[34]};
  const float* br_b[3] = {(const float*)d_in[29], (const float*)d_in[32], (const float*)d_in[35]};

  char* wsc = (char*)d_ws;
  float* stats = (float*)wsc;                 // [0:16) dw, [16:208) br, [208:272) gn2
  float* ss = (float*)(wsc + 2048);
  float* dwScale = ss;        float* dwShift = ss + 8;
  float* brScale = ss + 16;   float* brShift = ss + 112;
  float* gn2Scale = ss + 208; float* gn2Shift = ss + 304;

  size_t off = 4096;
  _Float16* yb = (_Float16*)(wsc + off);      off += (size_t)8 * NPIX * 2;    // 16MB, [pix][8]
  _Float16* zb = (_Float16*)(wsc + off);      off += (size_t)96 * NPIX * 2;   // 192MB, [ch][pix]
  float* attnBias = (float*)(wsc + off);      off += (size_t)8 * NPIX * 4;    // 32MB
  _Float16* xn   = (_Float16*)(wsc + off);    off += 1024 * 256 * 2;
  _Float16* qb   = (_Float16*)(wsc + off);    off += 1024 * 256 * 2;
  _Float16* kbuf = (_Float16*)(wsc + off);    off += 1024 * 256 * 2;
  _Float16* vbuf = (_Float16*)(wsc + off);    off += 1024 * 256 * 2;
  _Float16* aOut = (_Float16*)(wsc + off);    off += 1024 * 256 * 2;
  float*    x1   = (float*)(wsc + off);       off += 1024 * 256 * 4;
  _Float16* xn2  = (_Float16*)(wsc + off);    off += 1024 * 256 * 2;
  _Float16* h1   = (_Float16*)(wsc + off);    off += 1024 * 512 * 2;

  hipMemsetAsync(stats, 0, 272 * sizeof(float), stream);

  // ---- pairwise CNN ----
  k_dwconv<<<NPIX / 256, 256, 0, stream>>>(xpw, dw_w, yb, stats);
  k_finalize<<<1, 128, 0, stream>>>(stats, dw_g, dw_b, dwScale, dwShift, 8, 1, 1.f / NPIX);
  k_gn_gelu<<<8 * NPIX / 256, 256, 0, stream>>>(yb, dwScale, dwShift, nullptr, 1, 0, 7);
  for (int br = 0; br < 3; ++br)
    k_brconv<<<16384, 256, 0, stream>>>(yb, br_w[br], zb, stats, br + 1, br);
  for (int br = 0; br < 3; ++br)
    k_finalize<<<1, 128, 0, stream>>>(stats + 16 + br * 64, br_g[br], br_b[br],
                                      brScale + br * 32, brShift + br * 32, 32, 1, 1.f / NPIX);
  k_gn_gelu<<<96 * NPIX / 256, 256, 0, stream>>>(zb, brScale, brShift, stats + 208, 3, 20, 127);
  k_finalize<<<1, 128, 0, stream>>>(stats + 208, post_g, post_b, gn2Scale, gn2Shift,
                                    96, 3, 1.f / (3.f * NPIX));
  k_head<<<NPIX / 256, 256, 0, stream>>>(zb, gn2Scale, gn2Shift, head_w, head_b, attnBias);

  // ---- attention ----
  k_layernorm<<<1024, 256, 0, stream>>>(x, ln1_g, ln1_b, xn);
  k_gemm16<<<256, 128, 0, stream>>>(xn, wq, bq, nullptr, nullptr, qb, 1024, 256, 256, 0);
  k_gemm16<<<256, 128, 0, stream>>>(xn, wk, bk, nullptr, nullptr, kbuf, 1024, 256, 256, 0);
  k_gemm16<<<256, 128, 0, stream>>>(xn, wv, bv, nullptr, nullptr, vbuf, 1024, 256, 256, 0);
  k_attn<<<512, 32, 0, stream>>>(qb, kbuf, vbuf, attnBias, gate, mask, aOut);
  k_gemm16<<<256, 128, 0, stream>>>(aOut, wo, bo, x, x1, nullptr, 1024, 256, 256, 0);

  // ---- FFN ----
  k_layernorm<<<1024, 256, 0, stream>>>(x1, ln2_g, ln2_b, xn2);
  k_gemm16<<<512, 128, 0, stream>>>(xn2, w1, b1, nullptr, nullptr, h1, 1024, 256, 512, 1);
  k_gemm16<<<256, 128, 0, stream>>>(h1, w2, b2, x1, (float*)d_out, nullptr, 1024, 512, 256, 0);

  // ---- second tuple output: x_pairwise passthrough ----
  hipMemcpyAsync((float*)d_out + 1024 * 256, d_in[1], (size_t)8 * NPIX * sizeof(float),
                 hipMemcpyDeviceToDevice, stream);
}
